// DAttentionBaseline_17463337025866
// MI455X (gfx1250) — matrix operand order repaired and checked by reading, not compiled, not run on hardware
//
#include <hip/hip_runtime.h>
#include <hip/hip_bf16.h>
#include <math.h>

// ---------------------------------------------------------------------------
// Problem constants (from reference): B=32, NC=512, H=32, W=16, G=4, HEADS=8
//   GC=128, HC=64, Hk=32, Wk=16, n = HW = 512
// ---------------------------------------------------------------------------
#define HW      512          // H*W = Hk*Wk = n
#define MAPSZ   (32*512*512) // B*NC*HW floats, one feature map / q / k / v / o
#define NBAT    128          // B*G
#define GCC     128          // GC

typedef __attribute__((ext_vector_type(16))) _Float16 v16h;
typedef __attribute__((ext_vector_type(8)))  _Float16 v8h;
typedef __attribute__((ext_vector_type(4)))  _Float16 v4h;
typedef __attribute__((ext_vector_type(8)))  float    v8f;

__device__ __forceinline__ v8f wmma_f16(v16h a, v16h b, v8f c) {
  // D = A(16x32 f16) * B(32x16 f16) + C(16x16 f32)
  return __builtin_amdgcn_wmma_f32_16x16x32_f16(
      /*neg_a=*/false, a, /*neg_b=*/false, b,
      /*c_mod=*/(short)0, c, /*reuse_a=*/false, /*reuse_b=*/false);
}

__device__ __forceinline__ v16h cat8(v8h lo, v8h hi) {
  return __builtin_shufflevector(lo, hi, 0, 1, 2, 3, 4, 5, 6, 7,
                                 8, 9, 10, 11, 12, 13, 14, 15);
}

__device__ __forceinline__ float gelu_f(float x) {
  return 0.5f * x * (1.0f + erff(x * 0.70710678118654752f));
}

// ---------------------------------------------------------------------------
// Generic batched WMMA GEMM: C[bat][m][n] = act( W[m][k] * X[bat][k][n] + bias[m] )
// Block tile: 128(M) x 64(N), 8 waves, wave = 16x64 via 4 WMMA accumulators.
// LDS layouts are fragment-native so every WMMA operand is 2x ds_load_b128:
//   As[row][k]  (A frag: lane=M row, halves = K per ISA split)
//   Bt[col][k]  (B tile TRANSPOSED: lane=N col, halves = K per ISA split)
// X addressing generalized with a K-block stride so the concatenated q_off
// input (3 maps x 128 ch) is consumed in place.
// ---------------------------------------------------------------------------
__global__ __launch_bounds__(256) void gemm_wmma_kernel(
    const float* __restrict__ Wm, const float* __restrict__ X,
    const float* __restrict__ bias, float* __restrict__ C,
    int Kdim, int Ntot,
    long long xBat, int xRow, int kShift, long long kBlk,
    long long cBat, int act)
{
  __shared__ _Float16 As[128 * 32]; // [row 0..127][k 0..31]
  __shared__ _Float16 Bt[64 * 32];  // [col 0..63][k 0..31]  (transposed)

  const int tid   = threadIdx.x;
  const int lane  = tid & 31;
  const int wave  = tid >> 5;
  const int halfs = lane >> 4;   // 0: lanes 0-15, 1: lanes 16-31
  const int l16   = lane & 15;
  const int n0    = blockIdx.x * 64;
  const int m0    = blockIdx.y * 128;
  const int bat   = blockIdx.z;
  const int kMask = (1 << kShift) - 1;

  const float* Xb = X + (size_t)bat * (size_t)xBat;

  v8f acc[4] = {};

  for (int kk = 0; kk < Kdim; kk += 32) {
    // --- A tile (128x32): float4 global loads, v8h LDS stores ---
    {
      const int row = tid >> 1;            // 0..127
      const int kp  = (tid & 1) * 16;      // 0 or 16
      const float4* wp =
          (const float4*)(Wm + (size_t)(m0 + row) * Kdim + kk + kp);
      float4 f0 = wp[0], f1 = wp[1], f2 = wp[2], f3 = wp[3];
      v8h a0 = { (_Float16)f0.x, (_Float16)f0.y, (_Float16)f0.z, (_Float16)f0.w,
                 (_Float16)f1.x, (_Float16)f1.y, (_Float16)f1.z, (_Float16)f1.w };
      v8h a1 = { (_Float16)f2.x, (_Float16)f2.y, (_Float16)f2.z, (_Float16)f2.w,
                 (_Float16)f3.x, (_Float16)f3.y, (_Float16)f3.z, (_Float16)f3.w };
      *(v8h*)&As[row * 32 + kp]     = a0;
      *(v8h*)&As[row * 32 + kp + 8] = a1;
    }
    // --- B tile (32x64) stored transposed: per-K coalesced global reads ---
    {
      const int col = tid & 63;
      const int kg  = tid >> 6;            // 0..3 -> k = kg*8 .. kg*8+7
      v8h b;
      #pragma unroll
      for (int h = 0; h < 8; ++h) {
        int kq = kk + kg * 8 + h;
        b[h] = (_Float16)Xb[(size_t)(kq >> kShift) * (size_t)kBlk
                            + (size_t)(kq & kMask) * xRow + n0 + col];
      }
      *(v8h*)&Bt[col * 32 + kg * 8] = b;
    }
    __syncthreads();

    // --- A fragment: two contiguous v8h loads (ds_load_b128) ---
    const int arow = wave * 16 + l16;
    v16h af = cat8(*(const v8h*)&As[arow * 32 + halfs * 8],
                   *(const v8h*)&As[arow * 32 + 16 + halfs * 8]);
    // --- 4 B fragments / WMMAs: each two contiguous v8h loads ---
    #pragma unroll
    for (int j = 0; j < 4; ++j) {
      const int col = j * 16 + l16;
      v16h bf = cat8(*(const v8h*)&Bt[col * 32 + halfs * 8],
                     *(const v8h*)&Bt[col * 32 + 16 + halfs * 8]);
      acc[j] = wmma_f16(af, bf, acc[j]);
    }
    __syncthreads();
  }

  // --- epilogue: bias + optional GELU, fp32 store (C layout per ISA) ---
  #pragma unroll
  for (int r = 0; r < 8; ++r) {
    int m = m0 + wave * 16 + r + 8 * halfs;
    float bv = bias ? bias[m] : 0.0f;
    #pragma unroll
    for (int j = 0; j < 4; ++j) {
      float v = acc[j][r] + bv;
      if (act) v = gelu_f(v);
      C[(size_t)bat * (size_t)cBat + (size_t)m * Ntot + n0 + j * 16 + l16] = v;
    }
  }
}

// ---------------------------------------------------------------------------
// Offset tail: per (bat, s) position, reduce 128 channels of t (already GELU'd
// by the co1 GEMM epilogue) through the 1x1 depthwise conv (scale+bias), GELU,
// the 2xGC co3 projection, tanh scaling, add reference points -> grid (x,y).
// ---------------------------------------------------------------------------
__global__ void offset_grid_kernel(const float* __restrict__ t,
                                   const float* __restrict__ co2w,
                                   const float* __restrict__ co2b,
                                   const float* __restrict__ co3w,
                                   float* __restrict__ grid)
{
  int idx = blockIdx.x * blockDim.x + threadIdx.x;   // over 128*512
  if (idx >= NBAT * HW) return;
  int bat = idx >> 9;
  int s   = idx & 511;
  const float* tb = t + (size_t)bat * GCC * HW + s;
  float a0 = 0.f, a1 = 0.f;
  for (int c = 0; c < GCC; ++c) {
    float u = tb[(size_t)c * HW];
    u = u * co2w[c] + co2b[c];       // depthwise 1x1 conv
    u = gelu_f(u);
    a0 += co3w[c]       * u;         // off channel 0 (y)
    a1 += co3w[GCC + c] * u;         // off channel 1 (x)
  }
  float offy = tanhf(a0) * (2.0f / 31.0f);   // OFFSET_RANGE_FACTOR / (Hk-1)
  float offx = tanhf(a1) * (2.0f / 15.0f);   // OFFSET_RANGE_FACTOR / (Wk-1)
  int h = s >> 4, w = s & 15;
  float ry = (float)h * (2.0f / 31.0f) - 1.0f;
  float rx = (float)w * (2.0f / 15.0f) - 1.0f;
  grid[idx * 2]     = offx + rx;   // grid[...,0] = x
  grid[idx * 2 + 1] = offy + ry;   // grid[...,1] = y
}

// ---------------------------------------------------------------------------
// Bilinear grid sample with border-zero: one thread per (bat, c, p) output.
// img viewed as [B*G=128][GC=128][H=32][W=16]; xs contiguous -> [B][NC][n].
// ---------------------------------------------------------------------------
__global__ void grid_sample_kernel(const float* __restrict__ img,
                                   const float* __restrict__ grid,
                                   float* __restrict__ xs)
{
  int idx = blockIdx.x * blockDim.x + threadIdx.x;   // 128*128*512
  if (idx >= NBAT * GCC * HW) return;
  int p   = idx & 511;
  int c   = (idx >> 9) & 127;
  int bat = idx >> 16;
  float gx = grid[((size_t)bat * HW + p) * 2];
  float gy = grid[((size_t)bat * HW + p) * 2 + 1];
  float fx = (gx + 1.0f) * 0.5f * 15.0f;
  float fy = (gy + 1.0f) * 0.5f * 31.0f;
  float x0 = floorf(fx), y0 = floorf(fy);
  float wx1 = fx - x0, wx0 = 1.0f - wx1;
  float wy1 = fy - y0, wy0 = 1.0f - wy1;
  const float* im = img + ((size_t)bat * GCC + c) * HW;

  auto tap = [&](float xi, float yi) -> float {
    bool valid = (xi >= 0.0f) & (xi <= 15.0f) & (yi >= 0.0f) & (yi <= 31.0f);
    int xc = min(max((int)xi, 0), 15);
    int yc = min(max((int)yi, 0), 31);
    float v = im[yc * 16 + xc];
    return valid ? v : 0.0f;
  };
  xs[idx] = tap(x0,        y0       ) * wx0 * wy0
          + tap(x0 + 1.0f, y0       ) * wx1 * wy0
          + tap(x0,        y0 + 1.0f) * wx0 * wy1
          + tap(x0 + 1.0f, y0 + 1.0f) * wx1 * wy1;
}

// ---------------------------------------------------------------------------
// Flash-style attention for one (b, head): q,k,v are [64][512] f32 slices.
//   S = (scale*q)^T k  via WMMA (K=64 -> 2 k-steps)
//   online softmax over n, O += P * V^T via WMMA (P transposed through LDS)
// Block: 128 threads (4 waves).
// LDS: K stored TRANSPOSED [n][c] so S B-fragments are contiguous v8h pairs;
//      V stored [c][n] (its fragment is contiguous over n already);
//      per-wave P bounce buffer for the C->A layout transpose.
// ---------------------------------------------------------------------------
__global__ __launch_bounds__(128) void attn_kernel(
    const float* __restrict__ q, const float* __restrict__ k,
    const float* __restrict__ v, float* __restrict__ o)
{
  extern __shared__ _Float16 smem[];
  _Float16* ldkT = smem;                 // [n 0..511][c 0..63]
  _Float16* ldv  = ldkT + HW * 64;       // [c 0..63][n 0..511]
  _Float16* ldp  = ldv + 64 * HW;        // 4 waves x [16][32]

  const int bh    = blockIdx.x;          // b*HEADS + head
  const int tid   = threadIdx.x;
  const int lane  = tid & 31;
  const int wave  = tid >> 5;
  const int halfs = lane >> 4;
  const int l16   = lane & 15;
  const float scale = 0.125f;            // HC^-0.5

  const float* qb = q + (size_t)bh * 64 * HW;
  const float* kb = k + (size_t)bh * 64 * HW;
  const float* vb = v + (size_t)bh * 64 * HW;
  float*       ob = o + (size_t)bh * 64 * HW;

  // K -> LDS transposed [n][c]: v8h stores, per-c global reads
  {
    const int c0 = (tid & 7) * 8;        // channel group
    #pragma unroll 1
    for (int it = 0; it < 32; ++it) {
      int n = (tid >> 3) + it * 16;
      v8h t;
      #pragma unroll
      for (int h = 0; h < 8; ++h)
        t[h] = (_Float16)kb[(size_t)(c0 + h) * HW + n];
      *(v8h*)&ldkT[n * 64 + c0] = t;
    }
  }
  // V -> LDS [c][n]: float4 loads, v4h stores
  for (int i = tid * 4; i < 64 * HW; i += 128 * 4) {
    float4 f = *(const float4*)(vb + i);
    v4h t = { (_Float16)f.x, (_Float16)f.y, (_Float16)f.z, (_Float16)f.w };
    *(v4h*)&ldv[i] = t;
  }
  __syncthreads();

  _Float16* myp = ldp + wave * (16 * 32);

  for (int mt = wave; mt < 32; mt += 4) {        // 16-row query tiles
    const int m0 = mt * 16;

    // Q A-fragments (scale folded in), two K=32 steps covering c=0..63
    v16h aq[2];
    #pragma unroll
    for (int ks = 0; ks < 2; ++ks) {
      #pragma unroll
      for (int h = 0; h < 8; ++h) {
        int cA = ks * 32 + halfs * 8 + h;
        int cB = ks * 32 + 16 + halfs * 8 + h;
        aq[ks][h]     = (_Float16)(qb[(size_t)cA * HW + m0 + l16] * scale);
        aq[ks][h + 8] = (_Float16)(qb[(size_t)cB * HW + m0 + l16] * scale);
      }
    }

    v8f oacc[4] = {};
    float rowmax[8], rowsum[8];
    #pragma unroll
    for (int r = 0; r < 8; ++r) { rowmax[r] = -1.0e30f; rowsum[r] = 0.0f; }

    for (int nt = 0; nt < 16; ++nt) {            // 32-key chunks
      const int n0 = nt * 32;

      // S chunk: two 16x16 accumulators over K=64 channels
      v8f s0 = {}, s1 = {};
      #pragma unroll
      for (int ks = 0; ks < 2; ++ks) {
        const int crow = ks * 32 + halfs * 8;    // K (channel) base for this lane
        v16h bk0 = cat8(*(const v8h*)&ldkT[(n0 + l16) * 64 + crow],
                        *(const v8h*)&ldkT[(n0 + l16) * 64 + crow + 16]);
        v16h bk1 = cat8(*(const v8h*)&ldkT[(n0 + 16 + l16) * 64 + crow],
                        *(const v8h*)&ldkT[(n0 + 16 + l16) * 64 + crow + 16]);
        s0 = wmma_f16(aq[ks], bk0, s0);
        s1 = wmma_f16(aq[ks], bk1, s1);
      }

      // online softmax update (rows live across 16-lane halves)
      float p0[8], p1[8];
      #pragma unroll
      for (int r = 0; r < 8; ++r) {
        float cm = fmaxf(s0[r], s1[r]);
        cm = fmaxf(cm, __shfl_xor(cm, 1, 32));
        cm = fmaxf(cm, __shfl_xor(cm, 2, 32));
        cm = fmaxf(cm, __shfl_xor(cm, 4, 32));
        cm = fmaxf(cm, __shfl_xor(cm, 8, 32));
        float nm    = fmaxf(rowmax[r], cm);
        float alpha = __expf(rowmax[r] - nm);
        rowmax[r]   = nm;
        p0[r] = __expf(s0[r] - nm);
        p1[r] = __expf(s1[r] - nm);
        float cs = p0[r] + p1[r];
        cs += __shfl_xor(cs, 1, 32);
        cs += __shfl_xor(cs, 2, 32);
        cs += __shfl_xor(cs, 4, 32);
        cs += __shfl_xor(cs, 8, 32);
        rowsum[r] = rowsum[r] * alpha + cs;
        #pragma unroll
        for (int j = 0; j < 4; ++j) oacc[j][r] *= alpha;
      }

      // bounce P (C layout -> A layout transpose) through per-wave LDS
      #pragma unroll
      for (int r = 0; r < 8; ++r) {
        int mrow = r + 8 * halfs;
        myp[mrow * 32 + l16]      = (_Float16)p0[r];
        myp[mrow * 32 + 16 + l16] = (_Float16)p1[r];
      }
      asm volatile("s_wait_dscnt 0" ::: "memory");  // wave-local LDS RAW

      v16h ap = cat8(*(const v8h*)&myp[l16 * 32 + halfs * 8],
                     *(const v8h*)&myp[l16 * 32 + 16 + halfs * 8]);
      // O += P(16x32) * V^T(32x64) : 4 WMMAs over channel groups
      #pragma unroll
      for (int j = 0; j < 4; ++j) {
        const int c = j * 16 + l16;
        v16h bv = cat8(*(const v8h*)&ldv[(size_t)c * HW + n0 + halfs * 8],
                       *(const v8h*)&ldv[(size_t)c * HW + n0 + halfs * 8 + 16]);
        oacc[j] = wmma_f16(ap, bv, oacc[j]);
      }
    }

    // normalize and write O as [c][m] so the out-proj GEMM consumes it directly
    #pragma unroll
    for (int r = 0; r < 8; ++r) {
      float inv = 1.0f / rowsum[r];
      int mrow = m0 + r + 8 * halfs;
      #pragma unroll
      for (int j = 0; j < 4; ++j)
        ob[(size_t)(j * 16 + l16) * HW + mrow] = oacc[j][r] * inv;
    }
  }
}

// ---------------------------------------------------------------------------
// Host-side orchestration
// ---------------------------------------------------------------------------
static void launch_gemm(const float* Wm, const float* X, const float* bias, float* C,
                        int Kdim, int Mdim, int Ntot, int batches,
                        long long xBat, int xRow, int kShift, long long kBlk,
                        long long cBat, int act, hipStream_t stream)
{
  dim3 grid(Ntot / 64, Mdim / 128, batches);
  gemm_wmma_kernel<<<grid, 256, 0, stream>>>(Wm, X, bias, C, Kdim, Ntot,
                                             xBat, xRow, kShift, kBlk, cBat, act);
}

extern "C" void kernel_launch(void* const* d_in, const int* in_sizes, int n_in,
                              void* d_out, int out_size, void* d_ws, size_t ws_size,
                              hipStream_t stream) {
  const float* xin  = (const float*)d_in[0];
  const float* yin  = (const float*)d_in[1];
  const float* zin  = (const float*)d_in[2];
  const float* qw   = (const float*)d_in[3];
  const float* qbb  = (const float*)d_in[4];
  const float* kw   = (const float*)d_in[5];
  const float* kbb  = (const float*)d_in[6];
  const float* vw   = (const float*)d_in[7];
  const float* vbb  = (const float*)d_in[8];
  const float* ow   = (const float*)d_in[9];
  const float* obb  = (const float*)d_in[10];
  const float* co1w = (const float*)d_in[11];
  const float* co1b = (const float*)d_in[12];
  const float* co2w = (const float*)d_in[13];
  const float* co2b = (const float*)d_in[14];
  const float* co3w = (const float*)d_in[15];
  const float* maps[3] = { xin, yin, zin };

  float* ws     = (float*)d_ws;
  const long long MS = MAPSZ;            // 8,388,608 floats
  float* q_all  = ws;                    // 3 * MS  (q for x,y,z, contiguous)
  float* tbuf   = ws + 3 * MS;           // MS      (offset-net hidden, GELU'd)
  float* gridb  = ws + 4 * MS;           // 128*512*2
  float* xsbuf  = gridb + (long long)NBAT * HW * 2;   // MS (sampled features)
  float* kbuf   = xsbuf + MS;            // MS
  float* vbuf   = kbuf + MS;             // MS
  float* obuf   = vbuf + MS;             // MS
  float* outp   = (float*)d_out;

  (void)hipFuncSetAttribute((const void*)attn_kernel,
                            hipFuncAttributeMaxDynamicSharedMemorySize, 135168);

  // 1) q projections for all three maps (kept contiguous for co1's concat view)
  for (int m = 0; m < 3; ++m)
    launch_gemm(qw, maps[m], qbb, q_all + (long long)m * MS,
                /*K*/512, /*M*/512, /*N*/HW, /*bat*/32,
                /*xBat*/512LL * HW, /*xRow*/HW, /*kShift*/30, /*kBlk*/0,
                /*cBat*/512LL * HW, /*act*/0, stream);

  // 2) offset net co1 (+GELU): batches = B*G = 128, K = 3*GC = 384
  //    X[bat][k][n], k = map*128 + c -> q_all + map*MS + bat*65536 + c*512 + n
  launch_gemm(co1w, q_all, co1b, tbuf,
              /*K*/384, /*M*/128, /*N*/HW, /*bat*/NBAT,
              /*xBat*/(long long)GCC * HW, /*xRow*/HW, /*kShift*/7, /*kBlk*/MS,
              /*cBat*/(long long)GCC * HW, /*act*/1, stream);

  // 3) depthwise conv + GELU + co3 + tanh + reference points -> sampling grid
  offset_grid_kernel<<<(NBAT * HW + 255) / 256, 256, 0, stream>>>(
      tbuf, co2w, co2b, co3w, gridb);

  // 4) per map: sample -> k/v proj -> attention -> out proj
  for (int m = 0; m < 3; ++m) {
    const float* qm = q_all + (long long)m * MS;

    grid_sample_kernel<<<(NBAT * GCC * HW + 255) / 256, 256, 0, stream>>>(
        maps[m], gridb, xsbuf);

    launch_gemm(kw, xsbuf, kbb, kbuf, 512, 512, HW, 32,
                512LL * HW, HW, 30, 0, 512LL * HW, 0, stream);
    launch_gemm(vw, xsbuf, vbb, vbuf, 512, 512, HW, 32,
                512LL * HW, HW, 30, 0, 512LL * HW, 0, stream);

    attn_kernel<<<32 * 8, 128, 135168, stream>>>(qm, kbuf, vbuf, obuf);

    launch_gemm(ow, obuf, obb, outp + (long long)m * MS, 512, 512, HW, 32,
                512LL * HW, HW, 30, 0, 512LL * HW, 0, stream);
  }
}
